// MPNetSelfAttention_46566035423310
// MI455X (gfx1250) — matrix operand order repaired and checked by reading, not compiled, not run on hardware
//
#include <hip/hip_runtime.h>
#include <hip/hip_bf16.h>
#include <cstdint>

typedef __attribute__((ext_vector_type(16))) _Float16 v16h;
typedef __attribute__((ext_vector_type(8)))  _Float16 v8h;
typedef __attribute__((ext_vector_type(8)))  float    v8f;
typedef __attribute__((ext_vector_type(4)))  int      v4i;

constexpr int BB = 2;
constexpr int SS = 2048;
constexpr int DD = 1024;
constexpr int HH = 16;
constexpr int HD = 64;

// ---------------------------------------------------------------------------
// Async global->LDS copy (gfx1250 GLOBAL_LOAD_ASYNC_TO_LDS_B128, ASYNCcnt).
#if __has_builtin(__builtin_amdgcn_global_load_async_to_lds_b128)
#define HAVE_ASYNC_LDS 1
#else
#define HAVE_ASYNC_LDS 0
#endif

typedef v4i __attribute__((address_space(1)))* gv4i_p;  // global int4*
typedef v4i __attribute__((address_space(3)))* lv4i_p;  // LDS int4*

__device__ __forceinline__ void cp16(_Float16* l, const _Float16* g) {
#if HAVE_ASYNC_LDS
  __builtin_amdgcn_global_load_async_to_lds_b128(
      (gv4i_p)(uintptr_t)g, (lv4i_p)(uint32_t)(uintptr_t)l, 0, 0);
#else
  *(v8h*)l = *(const v8h*)g;
#endif
}

template <int N>
__device__ __forceinline__ void wait_async() {
#if HAVE_ASYNC_LDS
#if __has_builtin(__builtin_amdgcn_s_wait_asynccnt)
  __builtin_amdgcn_s_wait_asynccnt(N);
#else
  asm volatile("s_wait_asynccnt %0" ::"i"(N) : "memory");
#endif
#endif
}

// ---------------------------------------------------------------------------
// WMMA fragment helpers (wave32, v_wmma_f32_16x16x32_f16)
// A operand 16x32 (MxK): lanes 0-15: M=lane, f[0..7]=K0..7, f[8..15]=K16..23;
//                        lanes16-31: M=lane-16, K8..15 / K24..31.
__device__ __forceinline__ v16h frag_a(const _Float16* tile, int ld) {
  const int lane = threadIdx.x & 31;
  const int m  = lane & 15;
  const int kb = (lane >> 4) << 3;
  const _Float16* p = tile + m * ld + kb;
  v8h lo = *(const v8h*)(p);
  v8h hi = *(const v8h*)(p + 16);
  v16h f;
#pragma unroll
  for (int i = 0; i < 8; ++i) { f[i] = lo[i]; f[i + 8] = hi[i]; }
  return f;
}

// B operand 32x16 (KxN) given as B^T tile [N][K] row-major:
// lanes 0-15: N=lane, K0..7 / K16..23; lanes 16-31: N=lane-16, K8..15 / K24..31.
__device__ __forceinline__ v16h frag_b(const _Float16* bt, int ld) {
  const int lane = threadIdx.x & 31;
  const int n  = lane & 15;
  const int kb = (lane >> 4) << 3;
  const _Float16* p = bt + n * ld + kb;
  v8h lo = *(const v8h*)(p);
  v8h hi = *(const v8h*)(p + 16);
  v16h f;
#pragma unroll
  for (int i = 0; i < 8; ++i) { f[i] = lo[i]; f[i + 8] = hi[i]; }
  return f;
}

__device__ __forceinline__ v8f wmma_f16(v16h a, v16h b, v8f c) {
  return __builtin_amdgcn_wmma_f32_16x16x32_f16(false, a, false, b, (short)0, c,
                                                false, false);
}

// ---------------------------------------------------------------------------
// Pre-pass 1: hidden fp32 -> f16 (elementwise, vectorized)
__global__ __launch_bounds__(256) void cvt_hidden(const float* __restrict__ in,
                                                  _Float16* __restrict__ out) {
  size_t idx = ((size_t)blockIdx.x * 256 + threadIdx.x) * 8;
  float4 a = *(const float4*)(in + idx);
  float4 b = *(const float4*)(in + idx + 4);
  v8h o;
  o[0] = (_Float16)a.x; o[1] = (_Float16)a.y;
  o[2] = (_Float16)a.z; o[3] = (_Float16)a.w;
  o[4] = (_Float16)b.x; o[5] = (_Float16)b.y;
  o[6] = (_Float16)b.z; o[7] = (_Float16)b.w;
  *(v8h*)(out + idx) = o;
}

// Pre-pass 2: W fp32 [k][n] -> W^T f16 [n][k], LDS-tiled 32x32
__global__ __launch_bounds__(256) void wtrans(
    const float* __restrict__ W0, const float* __restrict__ W1,
    const float* __restrict__ W2, const float* __restrict__ W3,
    _Float16* __restrict__ T0, _Float16* __restrict__ T1,
    _Float16* __restrict__ T2, _Float16* __restrict__ T3) {
  const int z = blockIdx.z;
  const float* W = (z == 0) ? W0 : (z == 1) ? W1 : (z == 2) ? W2 : W3;
  _Float16* T    = (z == 0) ? T0 : (z == 1) ? T1 : (z == 2) ? T2 : T3;
  __shared__ _Float16 tile[32][33];
  const int k0 = blockIdx.x * 32, n0 = blockIdx.y * 32;
  const int t = threadIdx.x;
#pragma unroll
  for (int e = 0; e < 4; ++e) {
    int idx = e * 256 + t;
    int i = idx >> 5, j = idx & 31;
    tile[i][j] = (_Float16)W[(size_t)(k0 + i) * DD + n0 + j];
  }
  __syncthreads();
#pragma unroll
  for (int e = 0; e < 4; ++e) {
    int idx = e * 256 + t;
    int i = idx >> 5, j = idx & 31;
    T[(size_t)(n0 + i) * DD + k0 + j] = tile[j][i];
  }
}

// ---------------------------------------------------------------------------
// Generic f16 GEMM: A[4096 x 1024] @ B (given as B^T f16 [n][k]) + bias.
// Block tile 64(M) x 128(N), K-step 64; double-buffered async LDS pipeline.
// MODE 0: Q -> [B,H,S,HD] f16, scaled by 1/8     MODE 1: K -> [B,H,S,HD] f16
// MODE 2: V -> [B,H,HD,S] f16 (pre-transposed)   MODE 3: O -> fp32 [M,D]
template <int MODE>
__global__ __launch_bounds__(256) void gemm16(
    const _Float16* __restrict__ A, const _Float16* __restrict__ Bt,
    const float* __restrict__ bias, void* __restrict__ outp) {
  const int m0 = blockIdx.x * 64;
  const int n0 = blockIdx.y * 128;
  const int t = threadIdx.x;
  const int wid = t >> 5, lane = t & 31;
  const int wm = wid & 3, wn = wid >> 2;

  __shared__ _Float16 Ab[2][64 * 72];    // [m][k] tile
  __shared__ _Float16 Bb[2][128 * 72];   // [n][k] tile (pre-transposed)

  // issue 6 async 16B copies per thread for one K-step
  auto fill = [&](int k0, int bufi) {
    {
      int i = t >> 2, c = (t & 3) * 16;
      const _Float16* g = A + (size_t)(m0 + i) * DD + k0 + c;
      cp16(Ab[bufi] + i * 72 + c, g);
      cp16(Ab[bufi] + i * 72 + c + 8, g + 8);
    }
    {
      int i = t >> 1, c = (t & 1) * 32;
      const _Float16* g = Bt + (size_t)(n0 + i) * DD + k0 + c;
#pragma unroll
      for (int q = 0; q < 4; ++q) cp16(Bb[bufi] + i * 72 + c + q * 8, g + q * 8);
    }
  };

  v8f acc[4] = {};
  fill(0, 0);

  for (int k0 = 0; k0 < DD; k0 += 64) {
    const int cur = (k0 >> 6) & 1;
    __syncthreads();  // prior readers of buf[cur^1] are done
    if (k0 + 64 < DD) {
      fill(k0 + 64, cur ^ 1);
      wait_async<6>();  // async retires in order: current tile complete
    } else {
      wait_async<0>();
    }
    __syncthreads();  // all waves' fills of buf[cur] visible

    v16h a0 = frag_a(Ab[cur] + wm * 16 * 72, 72);
    v16h a1 = frag_a(Ab[cur] + wm * 16 * 72 + 32, 72);
#pragma unroll
    for (int j = 0; j < 4; ++j) {
      const _Float16* bt = Bb[cur] + (wn * 64 + j * 16) * 72;
      acc[j] = wmma_f16(a0, frag_b(bt, 72), acc[j]);
      acc[j] = wmma_f16(a1, frag_b(bt + 32, 72), acc[j]);
    }
  }

  const int hf = lane >> 4, nn = lane & 15;
#pragma unroll
  for (int j = 0; j < 4; ++j) {
    int gn = n0 + wn * 64 + j * 16 + nn;
    float bv = bias[gn];
#pragma unroll
    for (int r = 0; r < 8; ++r) {
      int gm = m0 + wm * 16 + r + hf * 8;
      float v = acc[j][r] + bv;
      if (MODE == 3) {
        ((float*)outp)[(size_t)gm * DD + gn] = v;
      } else {
        int b = gm >> 11, s = gm & 2047;
        int h = gn >> 6, hd = gn & 63;
        _Float16* o = (_Float16*)outp;
        if (MODE == 0) v *= 0.125f;  // 1/sqrt(HD) folded into Q
        if (MODE == 2)
          o[(size_t)((b * HH + h) * HD + hd) * SS + s] = (_Float16)v;
        else
          o[(size_t)((b * HH + h) * SS + s) * HD + hd] = (_Float16)v;
      }
    }
  }
}

// ---------------------------------------------------------------------------
// Flash attention: one block per (b, h, 128-row q tile); wave w owns query
// rows [w*16, w*16+16) so the online softmax never leaves the wave.
// K/V tiles double-buffered through the async global->LDS engine.
__global__ __launch_bounds__(256) void attn_kernel(
    const _Float16* __restrict__ Qh, const _Float16* __restrict__ Kh,
    const _Float16* __restrict__ Vh, const float* __restrict__ pbias,
    const float* __restrict__ mask, _Float16* __restrict__ ctx) {
  const int q0 = blockIdx.x * 128;
  const int bh = blockIdx.y;
  const int b = bh / HH, h = bh % HH;
  const int t = threadIdx.x, wid = t >> 5, lane = t & 31;
  const int hf = lane >> 4, nn = lane & 15;

  __shared__ _Float16 Kt[2][64 * 72];  // [k][hd]  (B^T for Q K^T)
  __shared__ _Float16 Vt[2][64 * 72];  // [hd][k]  (B^T for P V; V stored T)
  __shared__ _Float16 Pt[8][16 * 72];  // per-wave P scratch [16][64]

  const size_t hoff = (size_t)(b * HH + h) * SS * HD;
  const _Float16* Qbase = Qh + hoff + (size_t)(q0 + wid * 16) * HD;
  const _Float16* Kbase = Kh + hoff;
  const _Float16* Vbase = Vh + hoff;  // [HD][S] layout per head
  const float* brow = pbias + (size_t)h * SS * SS;
  const float* mrow = mask + (size_t)b * SS;

  // issue 4 async 16B copies per thread for one 64-key tile
  auto fill = [&](int kt, int bufi) {
    int i = t >> 2, c = (t & 3) * 16;
    const _Float16* gk = Kbase + (size_t)(kt + i) * HD + c;
    cp16(Kt[bufi] + i * 72 + c, gk);
    cp16(Kt[bufi] + i * 72 + c + 8, gk + 8);
    const _Float16* gv = Vbase + (size_t)i * SS + kt + c;
    cp16(Vt[bufi] + i * 72 + c, gv);
    cp16(Vt[bufi] + i * 72 + c + 8, gv + 8);
  };

  v16h qf0 = frag_a(Qbase, HD);
  v16h qf1 = frag_a(Qbase + 32, HD);

  v8f acc[4] = {};
  float rmax[8], rsum[8];
#pragma unroll
  for (int r = 0; r < 8; ++r) { rmax[r] = -1e30f; rsum[r] = 0.f; }

  fill(0, 0);

  for (int kt = 0; kt < SS; kt += 64) {
    const int cur = (kt >> 6) & 1;
    __syncthreads();  // prior readers of buf[cur^1] are done
    if (kt + 64 < SS) {
      fill(kt + 64, cur ^ 1);
      wait_async<4>();  // in-order retirement: current tile complete
    } else {
      wait_async<0>();
    }
    __syncthreads();  // all waves' fills of buf[cur] visible

    // scores = Q K^T (16 rows x 64 cols per wave)
    v8f s[4] = {};
#pragma unroll
    for (int j = 0; j < 4; ++j) {
      v16h kf0 = frag_b(Kt[cur] + (j * 16) * 72, 72);
      v16h kf1 = frag_b(Kt[cur] + (j * 16) * 72 + 32, 72);
      s[j] = wmma_f16(qf0, kf0, s[j]);
      s[j] = wmma_f16(qf1, kf1, s[j]);
    }

    // + position_bias (268 MB stream -> non-temporal) + attention_mask
#pragma unroll
    for (int j = 0; j < 4; ++j) {
      int kc = kt + j * 16 + nn;
      float mk = mrow[kc];
#pragma unroll
      for (int r = 0; r < 8; ++r) {
        int qm = q0 + wid * 16 + r + hf * 8;
        float pb = __builtin_nontemporal_load(&brow[(size_t)qm * SS + kc]);
        s[j][r] += pb + mk;
      }
    }

    // online softmax; rows live in 16-lane halves -> shfl_xor 1/2/4/8
    float tmax[8];
#pragma unroll
    for (int r = 0; r < 8; ++r) {
      tmax[r] = s[0][r];
#pragma unroll
      for (int j = 1; j < 4; ++j) tmax[r] = fmaxf(tmax[r], s[j][r]);
    }
#pragma unroll
    for (int off = 1; off < 16; off <<= 1)
#pragma unroll
      for (int r = 0; r < 8; ++r)
        tmax[r] = fmaxf(tmax[r], __shfl_xor(tmax[r], off, 32));

    float al[8];
#pragma unroll
    for (int r = 0; r < 8; ++r) {
      float mn = fmaxf(rmax[r], tmax[r]);
      al[r] = __expf(rmax[r] - mn);
      rmax[r] = mn;
    }
    float ps[8];
#pragma unroll
    for (int r = 0; r < 8; ++r) ps[r] = 0.f;
#pragma unroll
    for (int j = 0; j < 4; ++j)
#pragma unroll
      for (int r = 0; r < 8; ++r) {
        float p = __expf(s[j][r] - rmax[r]);
        s[j][r] = p;
        ps[r] += p;
      }
#pragma unroll
    for (int off = 1; off < 16; off <<= 1)
#pragma unroll
      for (int r = 0; r < 8; ++r) ps[r] += __shfl_xor(ps[r], off, 32);
#pragma unroll
    for (int r = 0; r < 8; ++r) rsum[r] = rsum[r] * al[r] + ps[r];
#pragma unroll
    for (int j = 0; j < 4; ++j)
#pragma unroll
      for (int r = 0; r < 8; ++r) acc[j][r] *= al[r];

    // P: C/D register layout -> f16 LDS -> A-operand layout
    _Float16* Pw = Pt[wid];
#pragma unroll
    for (int j = 0; j < 4; ++j)
#pragma unroll
      for (int r = 0; r < 8; ++r)
        Pw[(r + hf * 8) * 72 + j * 16 + nn] = (_Float16)s[j][r];
    __syncthreads();

    v16h pf0 = frag_a(Pw, 72);
    v16h pf1 = frag_a(Pw + 32, 72);
#pragma unroll
    for (int j2 = 0; j2 < 4; ++j2) {
      v16h vf0 = frag_b(Vt[cur] + (j2 * 16) * 72, 72);
      v16h vf1 = frag_b(Vt[cur] + (j2 * 16) * 72 + 32, 72);
      acc[j2] = wmma_f16(pf0, vf0, acc[j2]);
      acc[j2] = wmma_f16(pf1, vf1, acc[j2]);
    }
  }

#pragma unroll
  for (int j2 = 0; j2 < 4; ++j2)
#pragma unroll
    for (int r = 0; r < 8; ++r) {
      float v = acc[j2][r] / rsum[r];
      int qm = q0 + wid * 16 + r + hf * 8;
      ctx[(size_t)(b * SS + qm) * DD + h * HD + j2 * 16 + nn] = (_Float16)v;
    }
}

// ---------------------------------------------------------------------------
extern "C" void kernel_launch(void* const* d_in, const int* in_sizes, int n_in,
                              void* d_out, int out_size, void* d_ws,
                              size_t ws_size, hipStream_t stream) {
  (void)in_sizes; (void)n_in; (void)out_size; (void)ws_size;
  const float* hidden = (const float*)d_in[0];
  const float* mask   = (const float*)d_in[1];
  const float* pbias  = (const float*)d_in[2];
  const float* Wq = (const float*)d_in[3];
  const float* bq = (const float*)d_in[4];
  const float* Wk = (const float*)d_in[5];
  const float* bk = (const float*)d_in[6];
  const float* Wv = (const float*)d_in[7];
  const float* bv = (const float*)d_in[8];
  const float* Wo = (const float*)d_in[9];
  const float* bo = (const float*)d_in[10];
  float* out = (float*)d_out;

  const size_t NHID = (size_t)BB * SS * DD;  // 4 Mi
  const size_t NW = (size_t)DD * DD;         // 1 Mi
  const size_t NH = (size_t)BB * HH * SS * HD;
  _Float16* Hf  = (_Float16*)d_ws;
  _Float16* WqT = Hf + NHID;
  _Float16* WkT = WqT + NW;
  _Float16* WvT = WkT + NW;
  _Float16* WoT = WvT + NW;
  _Float16* Qd  = WoT + NW;
  _Float16* Kd  = Qd + NH;
  _Float16* Vd  = Kd + NH;
  _Float16* Cd  = Vd + NH;  // total 48 MB

  cvt_hidden<<<dim3(NHID / (256 * 8)), 256, 0, stream>>>(hidden, Hf);
  wtrans<<<dim3(32, 32, 4), 256, 0, stream>>>(Wq, Wk, Wv, Wo, WqT, WkT, WvT,
                                              WoT);
  dim3 gg((BB * SS) / 64, DD / 128);
  gemm16<0><<<gg, 256, 0, stream>>>(Hf, WqT, bq, (void*)Qd);
  gemm16<1><<<gg, 256, 0, stream>>>(Hf, WkT, bk, (void*)Kd);
  gemm16<2><<<gg, 256, 0, stream>>>(Hf, WvT, bv, (void*)Vd);
  attn_kernel<<<dim3(SS / 128, BB * HH), 256, 0, stream>>>(Qd, Kd, Vd, pbias,
                                                           mask, Cd);
  gemm16<3><<<gg, 256, 0, stream>>>(Cd, WoT, bo, (void*)out);
}
